// HomoConv_2920577761687
// MI455X (gfx1250) — matrix operand order repaired and checked by reading, not compiled, not run on hardware
//
#include <hip/hip_runtime.h>

typedef __attribute__((ext_vector_type(16))) __bf16 v16bf;
typedef __attribute__((ext_vector_type(8)))  __bf16 v8bf;
typedef __attribute__((ext_vector_type(8)))  float  v8f;
typedef __attribute__((ext_vector_type(4)))  float  v4f;

#define WMMA_BF16(A, B, C) \
  __builtin_amdgcn_wmma_f32_16x16x32_bf16(false, (A), false, (B), (short)0, (C), false, false)

// Load 8 contiguous fp32 and convert to 8 bf16 (one half of an A-fragment run).
__device__ __forceinline__ v8bf cvt8_f32_bf16(const float* __restrict__ p) {
  v4f a = *(const v4f*)p;
  v4f b = *(const v4f*)(p + 4);
  v8bf r;
  r[0] = (__bf16)a[0]; r[1] = (__bf16)a[1]; r[2] = (__bf16)a[2]; r[3] = (__bf16)a[3];
  r[4] = (__bf16)b[0]; r[5] = (__bf16)b[1]; r[6] = (__bf16)b[2]; r[7] = (__bf16)b[3];
  return r;
}

__device__ __forceinline__ v16bf afrag_join(v8bf lo, v8bf hi) {
  union { v16bf v; v8bf h[2]; } u;
  u.h[0] = lo; u.h[1] = hi;
  return u.v;
}

// Bias + LayerNorm (shuffle-reduce over the 16-lane half) + SiLU, result to LDS
// in A-fragment-friendly row-major bf16 [16][128].
__device__ __forceinline__ void bias_ln_silu_to_lds(
    v8f (&acc)[8], const float* __restrict__ b1, const float* __restrict__ g1,
    const float* __restrict__ bet1, __bf16* __restrict__ post, int m, int hh)
{
  float rs[8], rq[8];
  #pragma unroll
  for (int r = 0; r < 8; ++r) { rs[r] = 0.f; rq[r] = 0.f; }
  #pragma unroll
  for (int nt = 0; nt < 8; ++nt) {
    float bias = b1[nt * 16 + m];
    #pragma unroll
    for (int r = 0; r < 8; ++r) {
      float v = acc[nt][r] + bias;
      acc[nt][r] = v;
      rs[r] += v; rq[r] += v * v;
    }
  }
  #pragma unroll
  for (int mask = 1; mask < 16; mask <<= 1) {
    #pragma unroll
    for (int r = 0; r < 8; ++r) {
      rs[r] += __shfl_xor(rs[r], mask, 32);
      rq[r] += __shfl_xor(rq[r], mask, 32);
    }
  }
  float mu[8], rstd[8];
  #pragma unroll
  for (int r = 0; r < 8; ++r) {
    mu[r] = rs[r] * (1.0f / 128.0f);
    float var = rq[r] * (1.0f / 128.0f) - mu[r] * mu[r];
    rstd[r] = rsqrtf(var + 1e-5f);
  }
  #pragma unroll
  for (int nt = 0; nt < 8; ++nt) {
    float gg = g1[nt * 16 + m];
    float bb = bet1[nt * 16 + m];
    #pragma unroll
    for (int r = 0; r < 8; ++r) {
      float v = (acc[nt][r] - mu[r]) * rstd[r] * gg + bb;
      v = v / (1.0f + __expf(-v));                     // SiLU
      post[(r + 8 * hh) * 128 + nt * 16 + m] = (__bf16)v;
    }
  }
}

// Bias + residual + store one 16x128 C tile.
__device__ __forceinline__ void store_out_tile(
    const v8f (&acc2)[8], const float* __restrict__ b2, const float* __restrict__ resid,
    float* __restrict__ out, int base_row, int row_limit, int m, int hh)
{
  #pragma unroll
  for (int nt = 0; nt < 8; ++nt) {
    int col = nt * 16 + m;
    float bias = b2[col];
    #pragma unroll
    for (int r = 0; r < 8; ++r) {
      int orow = base_row + r + 8 * hh;
      if (orow < row_limit)
        out[(size_t)orow * 128 + col] = acc2[nt][r] + bias + resid[(size_t)orow * 128 + col];
    }
  }
}

// ---------------------------------------------------------------------------
// Prep: W[K][N=128] fp32 -> WT[N][K] bf16 (B-fragments become contiguous 32B)
// ---------------------------------------------------------------------------
__global__ void cvt_transpose_kernel(const float* __restrict__ W, __bf16* __restrict__ WT,
                                     int K, int N) {
  int idx = blockIdx.x * blockDim.x + threadIdx.x;
  if (idx < K * N) {
    int k = idx / N, n = idx - k * N;
    WT[(size_t)n * K + k] = (__bf16)W[idx];
  }
}

__global__ void zero_f32_kernel(float* __restrict__ p, long n) {
  long i = (long)blockIdx.x * blockDim.x + threadIdx.x;
  if (i < n) p[i] = 0.0f;
}

// ---------------------------------------------------------------------------
// Scatter-add edge features onto destination nodes (msgs is L2-resident).
// One thread handles 4 consecutive features of one edge -> coalesced atomics.
// ---------------------------------------------------------------------------
__global__ void scatter_add_kernel(const float* __restrict__ e, const int* __restrict__ dst,
                                   float* __restrict__ msgs, int E) {
  long gid = (long)blockIdx.x * blockDim.x + threadIdx.x;
  int j = (int)(gid >> 5);
  if (j >= E) return;
  int c = ((int)gid & 31) * 4;
  int d = dst[j];
  v4f v = *(const v4f*)(e + (size_t)j * 128 + c);
  float* base = msgs + (size_t)d * 128 + c;
  atomicAdd(base + 0, v[0]);
  atomicAdd(base + 1, v[1]);
  atomicAdd(base + 2, v[2]);
  atomicAdd(base + 3, v[3]);
}

// ---------------------------------------------------------------------------
// Node MLP: two 16-row tiles per wave (32 nodes) sharing every B-fragment.
// [32,256] x W1(256,128) -> bias -> LN -> SiLU -> x W2(128,128) -> bias -> +x
// ---------------------------------------------------------------------------
__global__ __launch_bounds__(128, 1) void node_mlp_kernel(
    const float* __restrict__ x, const float* __restrict__ msgs,
    const __bf16* __restrict__ W1T, const float* __restrict__ b1,
    const float* __restrict__ g1, const float* __restrict__ bet1,
    const __bf16* __restrict__ W2T, const float* __restrict__ b2,
    float* __restrict__ xout, int N)
{
  __shared__ __bf16 lds_post[4][2][16 * 128];   // 2 x 4KB per wave
  const int wave = threadIdx.x >> 5;
  const int lane = threadIdx.x & 31;
  const int m  = lane & 15;
  const int hh = lane >> 4;
  const int pairs = (N + 31) >> 5;
  const int pair = blockIdx.x * 4 + wave;
  if (pair >= pairs) return;
  const int base = pair * 32;
  int rowA = base + m;       if (rowA >= N) rowA = N - 1;
  int rowB = base + 16 + m;  if (rowB >= N) rowB = N - 1;
  const float* xrA = x    + (size_t)rowA * 128;
  const float* mrA = msgs + (size_t)rowA * 128;
  const float* xrB = x    + (size_t)rowB * 128;
  const float* mrB = msgs + (size_t)rowB * 128;

  // -------- GEMM1: K=256 (8 k-tiles), 8 n-tiles, 2 A-tiles per B-fragment ----
  v8f accA[8] = {}, accB[8] = {};
  #pragma unroll
  for (int kt = 0; kt < 8; ++kt) {
    int k0 = kt * 32 + hh * 16;
    int k1 = k0 + 8;
    const float* pA0 = (k0 < 128) ? (xrA + k0) : (mrA + (k0 - 128));
    const float* pA1 = (k1 < 128) ? (xrA + k1) : (mrA + (k1 - 128));
    const float* pB0 = (k0 < 128) ? (xrB + k0) : (mrB + (k0 - 128));
    const float* pB1 = (k1 < 128) ? (xrB + k1) : (mrB + (k1 - 128));
    v16bf aA = afrag_join(cvt8_f32_bf16(pA0), cvt8_f32_bf16(pA1));
    v16bf aB = afrag_join(cvt8_f32_bf16(pB0), cvt8_f32_bf16(pB1));
    #pragma unroll
    for (int nt = 0; nt < 8; ++nt) {
      v16bf bfr = *(const v16bf*)(W1T + (size_t)(nt * 16 + m) * 256 + kt * 32 + hh * 16);
      accA[nt] = WMMA_BF16(aA, bfr, accA[nt]);
      accB[nt] = WMMA_BF16(aB, bfr, accB[nt]);
    }
  }

  __bf16* postA = &lds_post[wave][0][0];
  __bf16* postB = &lds_post[wave][1][0];
  bias_ln_silu_to_lds(accA, b1, g1, bet1, postA, m, hh);
  bias_ln_silu_to_lds(accB, b1, g1, bet1, postB, m, hh);
  asm volatile("s_wait_dscnt 0" ::: "memory");   // wave-local LDS RAW fence

  // -------- GEMM2: K=128 (4 k-tiles), shared B-fragments --------
  v8f acc2A[8] = {}, acc2B[8] = {};
  #pragma unroll
  for (int kt = 0; kt < 4; ++kt) {
    int k0 = kt * 32 + hh * 16;
    v16bf aA = afrag_join(*(const v8bf*)(postA + m * 128 + k0),
                          *(const v8bf*)(postA + m * 128 + k0 + 8));
    v16bf aB = afrag_join(*(const v8bf*)(postB + m * 128 + k0),
                          *(const v8bf*)(postB + m * 128 + k0 + 8));
    #pragma unroll
    for (int nt = 0; nt < 8; ++nt) {
      v16bf bfr = *(const v16bf*)(W2T + (size_t)(nt * 16 + m) * 128 + kt * 32 + hh * 16);
      acc2A[nt] = WMMA_BF16(aA, bfr, acc2A[nt]);
      acc2B[nt] = WMMA_BF16(aB, bfr, acc2B[nt]);
    }
  }

  store_out_tile(acc2A, b2, x, xout, base, N, m, hh);
  store_out_tile(acc2B, b2, x, xout, base + 16, N, m, hh);
}

// ---------------------------------------------------------------------------
// Edge MLP: two 16-edge tiles per wave (32 edges) sharing every B-fragment.
// [32,384] (= x_out[start] | x_out[end] | e) x We1 -> bias -> LN -> SiLU
// -> x We2 -> bias -> +e residual.
// ---------------------------------------------------------------------------
__global__ __launch_bounds__(128, 1) void edge_mlp_kernel(
    const float* __restrict__ xout, const float* __restrict__ e,
    const int* __restrict__ startv, const int* __restrict__ endv,
    const __bf16* __restrict__ W1T, const float* __restrict__ b1,
    const float* __restrict__ g1, const float* __restrict__ bet1,
    const __bf16* __restrict__ W2T, const float* __restrict__ b2,
    float* __restrict__ eout, int E)
{
  __shared__ __bf16 lds_post[4][2][16 * 128];
  const int wave = threadIdx.x >> 5;
  const int lane = threadIdx.x & 31;
  const int m  = lane & 15;
  const int hh = lane >> 4;
  const int pairs = (E + 31) >> 5;
  const int pair = blockIdx.x * 4 + wave;
  if (pair >= pairs) return;
  const int base = pair * 32;
  int jA = base + m;       if (jA >= E) jA = E - 1;
  int jB = base + 16 + m;  if (jB >= E) jB = E - 1;
  const float* rA0 = xout + (size_t)startv[jA] * 128;
  const float* rA1 = xout + (size_t)endv[jA]   * 128;
  const float* rA2 = e    + (size_t)jA         * 128;
  const float* rB0 = xout + (size_t)startv[jB] * 128;
  const float* rB1 = xout + (size_t)endv[jB]   * 128;
  const float* rB2 = e    + (size_t)jB         * 128;

  // -------- GEMM1: K=384 (12 k-tiles), 8 n-tiles, 2 A-tiles per B-fragment --
  v8f accA[8] = {}, accB[8] = {};
  #pragma unroll
  for (int kt = 0; kt < 12; ++kt) {
    int k0 = kt * 32 + hh * 16;
    int k1 = k0 + 8;
    const float* pA0 = (k0 < 128) ? (rA0 + k0) : (k0 < 256) ? (rA1 + k0 - 128) : (rA2 + k0 - 256);
    const float* pA1 = (k1 < 128) ? (rA0 + k1) : (k1 < 256) ? (rA1 + k1 - 128) : (rA2 + k1 - 256);
    const float* pB0 = (k0 < 128) ? (rB0 + k0) : (k0 < 256) ? (rB1 + k0 - 128) : (rB2 + k0 - 256);
    const float* pB1 = (k1 < 128) ? (rB0 + k1) : (k1 < 256) ? (rB1 + k1 - 128) : (rB2 + k1 - 256);
    v16bf aA = afrag_join(cvt8_f32_bf16(pA0), cvt8_f32_bf16(pA1));
    v16bf aB = afrag_join(cvt8_f32_bf16(pB0), cvt8_f32_bf16(pB1));
    #pragma unroll
    for (int nt = 0; nt < 8; ++nt) {
      v16bf bfr = *(const v16bf*)(W1T + (size_t)(nt * 16 + m) * 384 + kt * 32 + hh * 16);
      accA[nt] = WMMA_BF16(aA, bfr, accA[nt]);
      accB[nt] = WMMA_BF16(aB, bfr, accB[nt]);
    }
  }

  __bf16* postA = &lds_post[wave][0][0];
  __bf16* postB = &lds_post[wave][1][0];
  bias_ln_silu_to_lds(accA, b1, g1, bet1, postA, m, hh);
  bias_ln_silu_to_lds(accB, b1, g1, bet1, postB, m, hh);
  asm volatile("s_wait_dscnt 0" ::: "memory");

  // -------- GEMM2: K=128 (4 k-tiles) --------
  v8f acc2A[8] = {}, acc2B[8] = {};
  #pragma unroll
  for (int kt = 0; kt < 4; ++kt) {
    int k0 = kt * 32 + hh * 16;
    v16bf aA = afrag_join(*(const v8bf*)(postA + m * 128 + k0),
                          *(const v8bf*)(postA + m * 128 + k0 + 8));
    v16bf aB = afrag_join(*(const v8bf*)(postB + m * 128 + k0),
                          *(const v8bf*)(postB + m * 128 + k0 + 8));
    #pragma unroll
    for (int nt = 0; nt < 8; ++nt) {
      v16bf bfr = *(const v16bf*)(W2T + (size_t)(nt * 16 + m) * 128 + kt * 32 + hh * 16);
      acc2A[nt] = WMMA_BF16(aA, bfr, acc2A[nt]);
      acc2B[nt] = WMMA_BF16(aB, bfr, acc2B[nt]);
    }
  }

  store_out_tile(acc2A, b2, e, eout, base, E, m, hh);
  store_out_tile(acc2B, b2, e, eout, base + 16, E, m, hh);
}

// ---------------------------------------------------------------------------
extern "C" void kernel_launch(void* const* d_in, const int* in_sizes, int n_in,
                              void* d_out, int out_size, void* d_ws, size_t ws_size,
                              hipStream_t stream) {
  (void)n_in; (void)out_size; (void)ws_size;
  const int H = 128;
  const float* x      = (const float*)d_in[0];
  const int*   eidx   = (const int*)  d_in[1];
  const float* e      = (const float*)d_in[2];
  const float* Wn1    = (const float*)d_in[3];
  const float* bn1    = (const float*)d_in[4];
  const float* gn1    = (const float*)d_in[5];
  const float* betan1 = (const float*)d_in[6];
  const float* Wn2    = (const float*)d_in[7];
  const float* bn2    = (const float*)d_in[8];
  const float* We1    = (const float*)d_in[9];
  const float* be1    = (const float*)d_in[10];
  const float* ge1    = (const float*)d_in[11];
  const float* betae1 = (const float*)d_in[12];
  const float* We2    = (const float*)d_in[13];
  const float* be2    = (const float*)d_in[14];

  const int N = in_sizes[0] / H;
  const int E = in_sizes[2] / H;

  float* xout = (float*)d_out;
  float* eout = (float*)d_out + (size_t)N * H;

  // Workspace layout (all 32B-aligned): msgs fp32, then bf16 transposed weights
  char* ws = (char*)d_ws;
  float* msgs = (float*)ws;
  size_t off = (size_t)N * H * sizeof(float);
  __bf16* Wn1T = (__bf16*)(ws + off); off += (size_t)2 * H * H * sizeof(__bf16);
  __bf16* Wn2T = (__bf16*)(ws + off); off += (size_t)H * H * sizeof(__bf16);
  __bf16* We1T = (__bf16*)(ws + off); off += (size_t)3 * H * H * sizeof(__bf16);
  __bf16* We2T = (__bf16*)(ws + off);

  // 1) weight prep (tiny)
  cvt_transpose_kernel<<<(2 * H * H + 255) / 256, 256, 0, stream>>>(Wn1, Wn1T, 2 * H, H);
  cvt_transpose_kernel<<<(H * H + 255) / 256, 256, 0, stream>>>(Wn2, Wn2T, H, H);
  cvt_transpose_kernel<<<(3 * H * H + 255) / 256, 256, 0, stream>>>(We1, We1T, 3 * H, H);
  cvt_transpose_kernel<<<(H * H + 255) / 256, 256, 0, stream>>>(We2, We2T, H, H);

  // 2) zero msgs, scatter-add e -> msgs[end]
  {
    long n = (long)N * H;
    zero_f32_kernel<<<(int)((n + 255) / 256), 256, 0, stream>>>(msgs, n);
    long t = (long)E * 32;   // 4 features per thread
    scatter_add_kernel<<<(int)((t + 255) / 256), 256, 0, stream>>>(e, eidx + E, msgs, E);
  }

  // 3) node MLP (32 nodes per wave)
  {
    int pairs = (N + 31) / 32;
    node_mlp_kernel<<<(pairs + 3) / 4, 128, 0, stream>>>(
        x, msgs, Wn1T, bn1, gn1, betan1, Wn2T, bn2, xout, N);
  }

  // 4) edge MLP (32 edges per wave; reads x_out written above; stream-ordered)
  {
    int pairs = (E + 31) / 32;
    edge_mlp_kernel<<<(pairs + 3) / 4, 128, 0, stream>>>(
        xout, e, eidx, eidx + E, We1T, be1, ge1, betae1, We2T, be2, eout, E);
  }
}
